// DSAM_mixer_37099927503040
// MI455X (gfx1250) — compile-verified, matrix operand order checked by reading, not run on hardware
//
#include <hip/hip_runtime.h>
#include <cstddef>

// ---------------------------------------------------------------------------
// DSAM / visual-Mamba mixer for MI455X (gfx1250, wave32, WMMA)
//   B=2, H=W=64, L=4096, d_model=192, d_inner=384, d_state=16, dt_rank=12
// ---------------------------------------------------------------------------

#define Bz      2
#define Hh      64
#define Ww      64
#define LSEQ    4096            // H*W
#define DMODEL  192
#define DINNER  384
#define DSTATE  16
#define DTRANK  12
#define XDBL_W  (DTRANK + 2 * DSTATE)   // 44
#define NROWS   (Bz * LSEQ)             // 8192

#define NCH     64              // chunks along L
#define CHL     64              // steps per chunk  (NCH*CHL == LSEQ)

typedef __attribute__((ext_vector_type(16))) __bf16 v16bf;
typedef __attribute__((ext_vector_type(8)))  float  v8f;

union FragBF { v16bf v; unsigned int u[8]; };

__device__ __forceinline__ unsigned short f32_to_bf16(float f) {
    union { float f; unsigned int u; } v; v.f = f;
    unsigned int lsb = (v.u >> 16) & 1u;
    return (unsigned short)((v.u + 0x7FFFu + lsb) >> 16);
}
__device__ __forceinline__ unsigned int pack_bf16(float lo, float hi) {
    return (unsigned int)f32_to_bf16(lo) | ((unsigned int)f32_to_bf16(hi) << 16);
}

__device__ __forceinline__ float sigmoidf_(float x) { return 1.f / (1.f + __expf(-x)); }
__device__ __forceinline__ float softplusf_(float x) {
    return (x > 20.f) ? x : log1pf(__expf(x));
}

// ---------------------------------------------------------------------------
// Generic WMMA GEMM:  C[M,N] = A[M,K] * W[N,K]^T   (fp32 memory, bf16 compute,
// f32 accumulate).  Block tile 128x64, 8 waves; each wave owns a 32x32 slab
// of C: 2 A-frags x 2 B-frags -> 4x v_wmma_f32_16x16x32_bf16 per K-step.
// Double-buffered LDS, one barrier per K-step.  Out-of-range rows are clamped
// (their products land only in discarded C entries), so the hot-loop staging
// is completely branch-free; only the ragged K tail takes a masked path, and
// that decision is a uniform scalar branch.
// ---------------------------------------------------------------------------
#define BM 128
#define BN 64
#define BK 32
#define LDT 36   // LDS row stride in ushorts: 72B rows -> 8B-aligned chunks

// fast path: 8 consecutive K-elements, unconditional 2x b128 loads
__device__ __forceinline__ void stage8_fast(unsigned short* dst,
                                            const float* __restrict__ src) {
    const float4* s4 = (const float4*)src;
    float4 a = s4[0];
    float4 b = s4[1];
    unsigned long long lo =
        (unsigned long long)pack_bf16(a.x, a.y) |
        ((unsigned long long)pack_bf16(a.z, a.w) << 32);
    unsigned long long hi =
        (unsigned long long)pack_bf16(b.x, b.y) |
        ((unsigned long long)pack_bf16(b.z, b.w) << 32);
    ((unsigned long long*)dst)[0] = lo;            // ds_store_b64 x2
    ((unsigned long long*)dst)[1] = hi;
}

// tail path: zero-fill K >= limit (only reached on the final ragged K tile)
__device__ __forceinline__ void stage8_tail(unsigned short* dst,
                                            const float* __restrict__ src,
                                            int gk, int K) {
    float v[8];
#pragma unroll
    for (int i = 0; i < 8; ++i) v[i] = (gk + i < K) ? src[i] : 0.f;
    unsigned long long lo =
        (unsigned long long)pack_bf16(v[0], v[1]) |
        ((unsigned long long)pack_bf16(v[2], v[3]) << 32);
    unsigned long long hi =
        (unsigned long long)pack_bf16(v[4], v[5]) |
        ((unsigned long long)pack_bf16(v[6], v[7]) << 32);
    ((unsigned long long*)dst)[0] = lo;
    ((unsigned long long*)dst)[1] = hi;
}

__device__ __forceinline__ void stage_tile(
    unsigned short (*Asb)[LDT], unsigned short (*Bsb)[LDT],
    const float* __restrict__ a0p, const float* __restrict__ a1p,
    const float* __restrict__ bp, int sr, int sk, int k0, int K)
{
    if (k0 + BK <= K) {          // uniform scalar branch
        stage8_fast(&Asb[sr][sk],      a0p + k0);
        stage8_fast(&Asb[sr + 64][sk], a1p + k0);
        stage8_fast(&Bsb[sr][sk],      bp  + k0);
    } else {
        stage8_tail(&Asb[sr][sk],      a0p + k0, k0 + sk, K);
        stage8_tail(&Asb[sr + 64][sk], a1p + k0, k0 + sk, K);
        stage8_tail(&Bsb[sr][sk],      bp  + k0, k0 + sk, K);
    }
}

__device__ __forceinline__ v16bf load_frag(const unsigned short (*S)[LDT],
                                           int rowbase, int lane) {
    // ISA 16-bit operand layout: lanes 0-15 -> rows, lanes 16-31 -> K+8 half;
    // VGPR v<4: K=2v..2v+1, v>=4: K=16+2(v-4)..  (pairs are LDS-contiguous)
    FragBF f;
    int row = rowbase + (lane & 15);
    int koff = (lane >> 4) * 8;
#pragma unroll
    for (int v = 0; v < 8; ++v) {
        int kp = (v < 4 ? 2 * v : 16 + 2 * (v - 4)) + koff;
        f.u[v] = *(const unsigned int*)&S[row][kp];
    }
    return f.v;
}

__device__ __forceinline__ void wmma_step(
    const unsigned short (*Asb)[LDT], const unsigned short (*Bsb)[LDT],
    int wm, int wn, int lane,
    v8f& acc00, v8f& acc01, v8f& acc10, v8f& acc11)
{
    v16bf a0 = load_frag(Asb, wm * 32,      lane);
    v16bf a1 = load_frag(Asb, wm * 32 + 16, lane);
    v16bf b0 = load_frag(Bsb, wn * 32,      lane);
    v16bf b1 = load_frag(Bsb, wn * 32 + 16, lane);
    acc00 = __builtin_amdgcn_wmma_f32_16x16x32_bf16(
                false, a0, false, b0, (short)0, acc00, false, false);
    acc01 = __builtin_amdgcn_wmma_f32_16x16x32_bf16(
                false, a0, false, b1, (short)0, acc01, false, false);
    acc10 = __builtin_amdgcn_wmma_f32_16x16x32_bf16(
                false, a1, false, b0, (short)0, acc10, false, false);
    acc11 = __builtin_amdgcn_wmma_f32_16x16x32_bf16(
                false, a1, false, b1, (short)0, acc11, false, false);
}

__global__ __launch_bounds__(256) void gemm_bf16_wmma(
    const float* __restrict__ A, int lda,
    const float* __restrict__ W, int ldb,
    float* __restrict__ C, int ldc,
    int M, int N, int K)
{
    __shared__ unsigned short As[2][BM][LDT];   // 2 x 9216 B
    __shared__ unsigned short Bs[2][BN][LDT];   // 2 x 4608 B

    const int tid  = threadIdx.x;
    const int lane = tid & 31;
    const int wid  = tid >> 5;
    const int wm   = wid & 3;        // 4 wave-tiles along M  (4*32 = 128)
    const int wn   = wid >> 2;       // 2 wave-tiles along N  (2*32 = 64)

    const int rowblk = blockIdx.y * BM;
    const int colblk = blockIdx.x * BN;

    v8f acc00 = {}, acc01 = {}, acc10 = {}, acc11 = {};

    // staging assignment: thread -> (row, 8-wide K chunk); rows clamped so
    // OOB rows read row 0 (their products only reach discarded C entries)
    const int sr = tid >> 2;            // 0..63
    const int sk = (tid & 3) * 8;       // 0,8,16,24

    const int grA0 = rowblk + sr;
    const int grA1 = rowblk + sr + 64;
    const int gcB  = colblk + sr;
    const float* __restrict__ a0p = A + (size_t)(grA0 < M ? grA0 : 0) * lda + sk;
    const float* __restrict__ a1p = A + (size_t)(grA1 < M ? grA1 : 0) * lda + sk;
    const float* __restrict__ bp  = W + (size_t)(gcB  < N ? gcB  : 0) * ldb + sk;

    // prologue: stage tile 0 into buffer 0
    stage_tile(As[0], Bs[0], a0p, a1p, bp, sr, sk, 0, K);
    __syncthreads();

    int p = 0;
    for (int k0 = 0; k0 < K; k0 += BK, p ^= 1) {
        int k1 = k0 + BK;
        if (k1 < K) {                    // stage next tile into other buffer
            if (k1 + BK < K) {           // hint tile k+2 (global_prefetch_b8)
                __builtin_prefetch(a0p + k1 + BK, 0, 1);
                __builtin_prefetch(bp  + k1 + BK, 0, 1);
            }
            stage_tile(As[p ^ 1], Bs[p ^ 1], a0p, a1p, bp, sr, sk, k1, K);
        }
        wmma_step(As[p], Bs[p], wm, wn, lane, acc00, acc01, acc10, acc11);
        __syncthreads();                 // one barrier per K-step
    }

    // ---- store: VGPR r -> row (lane<16 ? r : 8+r), col = lane&15 ----
    const int r0 = rowblk + wm * 32 + (lane >> 4) * 8;
    const int c0 = colblk + wn * 32 + (lane & 15);
#pragma unroll
    for (int r = 0; r < 8; ++r) {
        int rrA = r0 + r;
        int rrB = r0 + r + 16;
        if (rrA < M) {
            if (c0 < N)      C[(size_t)rrA * ldc + c0]      = acc00[r];
            if (c0 + 16 < N) C[(size_t)rrA * ldc + c0 + 16] = acc01[r];
        }
        if (rrB < M) {
            if (c0 < N)      C[(size_t)rrB * ldc + c0]      = acc10[r];
            if (c0 + 16 < N) C[(size_t)rrB * ldc + c0 + 16] = acc11[r];
        }
    }
}

// ---------------------------------------------------------------------------
// Depthwise 3x3 conv (SAME) + bias + SiLU.  Reads xi half of xz, writes x_seq.
// ---------------------------------------------------------------------------
__global__ __launch_bounds__(256) void conv_silu_kernel(
    const float* __restrict__ xz, const float* __restrict__ cw,
    const float* __restrict__ cb, float* __restrict__ xseq)
{
    int idx = blockIdx.x * 256 + threadIdx.x;       // over B*H*W*DINNER
    if (idx >= Bz * Hh * Ww * DINNER) return;
    int d = idx % DINNER; int rest = idx / DINNER;
    int w = rest % Ww; rest /= Ww;
    int h = rest % Hh; int b = rest / Hh;

    float acc = cb[d];
#pragma unroll
    for (int dh = -1; dh <= 1; ++dh) {
#pragma unroll
        for (int dw = -1; dw <= 1; ++dw) {
            int hh = h + dh, ww = w + dw;
            if (hh >= 0 && hh < Hh && ww >= 0 && ww < Ww) {
                int row = (b * Hh + hh) * Ww + ww;
                acc += xz[(size_t)row * (2 * DINNER) + d] *
                       cw[d * 9 + (dh + 1) * 3 + (dw + 1)];
            }
        }
    }
    xseq[idx] = acc * sigmoidf_(acc);
}

// ---------------------------------------------------------------------------
// Chunked selective scan.  h_l = dA_l * h_{l-1} + dBu_l  is affine -> per
// chunk aggregate (P = prod dA, q = chunk result from h=0), short serial
// prefix over 64 chunks, then parallel replay emitting y.
// flat index = ((b*DINNER + d) * NCH + chunk) * 16 + n
// ---------------------------------------------------------------------------
__global__ __launch_bounds__(256) void scan_chunk_kernel(
    const float* __restrict__ dtmm, const float* __restrict__ xseq,
    const float* __restrict__ xdbl, const float* __restrict__ dtb,
    const float* __restrict__ Alog,
    float* __restrict__ Pbuf, float* __restrict__ Qbuf)
{
    int flat = blockIdx.x * 256 + threadIdx.x;      // Bz*DINNER*NCH*16
    int n     = flat & 15;
    int chunk = (flat >> 4) & (NCH - 1);
    int bd    = flat >> 10;
    int d = bd % DINNER;
    int b = bd / DINNER;

    float acoef = -__expf(Alog[d * DSTATE + n]);
    float bias2 = 2.f * dtb[d];          // reference adds dt_proj_b twice
    float P = 1.f, q = 0.f;
    int l0 = b * LSEQ + chunk * CHL;
    for (int i = 0; i < CHL; ++i) {
        int row = l0 + i;
        float delta = softplusf_(dtmm[(size_t)row * DINNER + d] + bias2);
        float dA  = __expf(delta * acoef);
        float dBu = delta * xseq[(size_t)row * DINNER + d] *
                    xdbl[(size_t)row * XDBL_W + DTRANK + n];
        P *= dA;
        q  = dA * q + dBu;
    }
    Pbuf[flat] = P;
    Qbuf[flat] = q;
}

__global__ __launch_bounds__(256) void scan_prefix_kernel(
    const float* __restrict__ P, const float* __restrict__ Q,
    float* __restrict__ Hin)
{
    int t = blockIdx.x * 256 + threadIdx.x;         // Bz*DINNER*16
    if (t >= Bz * DINNER * DSTATE) return;
    int n  = t & 15;
    int bd = t >> 4;
    float h = 0.f;
    for (int c = 0; c < NCH; ++c) {
        int idx = ((bd * NCH + c) << 4) | n;
        Hin[idx] = h;
        h = P[idx] * h + Q[idx];
    }
}

__global__ __launch_bounds__(256) void scan_emit_kernel(
    const float* __restrict__ dtmm, const float* __restrict__ xseq,
    const float* __restrict__ xdbl, const float* __restrict__ pc,
    const float* __restrict__ dtb,  const float* __restrict__ Alog,
    const float* __restrict__ Dpv,  const float* __restrict__ Hin,
    float* __restrict__ y)
{
    int flat = blockIdx.x * 256 + threadIdx.x;
    int n     = flat & 15;
    int chunk = (flat >> 4) & (NCH - 1);
    int bd    = flat >> 10;
    int d = bd % DINNER;
    int b = bd / DINNER;

    float acoef = -__expf(Alog[d * DSTATE + n]);
    float bias2 = 2.f * dtb[d];
    float dpv   = Dpv[d];
    float h = Hin[flat];
    int l0 = b * LSEQ + chunk * CHL;
    for (int i = 0; i < CHL; ++i) {
        int row = l0 + i;
        float u     = xseq[(size_t)row * DINNER + d];
        float delta = softplusf_(dtmm[(size_t)row * DINNER + d] + bias2);
        float dA    = __expf(delta * acoef);
        float Bn    = xdbl[(size_t)row * XDBL_W + DTRANK + n];
        float Cn    = xdbl[(size_t)row * XDBL_W + DTRANK + DSTATE + n] +
                      pc[(size_t)row * DSTATE + n];
        h = dA * h + delta * u * Bn;
        // reduce h*Cn across the 16 state lanes of this channel (wave32)
        float contrib = h * Cn;
        contrib += __shfl_xor(contrib, 1, 32);
        contrib += __shfl_xor(contrib, 2, 32);
        contrib += __shfl_xor(contrib, 4, 32);
        contrib += __shfl_xor(contrib, 8, 32);
        if (n == 0)
            y[(size_t)row * DINNER + d] = contrib + dpv * u;
    }
}

// ---------------------------------------------------------------------------
// LayerNorm over d_inner + SiLU(z) gate.  One wave per row (12 elems/lane).
// ---------------------------------------------------------------------------
__global__ __launch_bounds__(256) void ln_gate_kernel(
    const float* __restrict__ y, const float* __restrict__ xz,
    const float* __restrict__ g, const float* __restrict__ bta,
    float* __restrict__ yg)
{
    int wid  = threadIdx.x >> 5;
    int lane = threadIdx.x & 31;
    int row  = blockIdx.x * 8 + wid;
    if (row >= NROWS) return;

    const float* yr = y + (size_t)row * DINNER;
    float vals[DINNER / 32];
    float s = 0.f, s2 = 0.f;
#pragma unroll
    for (int i = 0; i < DINNER / 32; ++i) {
        float v = yr[lane + i * 32];
        vals[i] = v; s += v; s2 += v * v;
    }
#pragma unroll
    for (int m = 16; m >= 1; m >>= 1) {
        s  += __shfl_xor(s,  m, 32);
        s2 += __shfl_xor(s2, m, 32);
    }
    float mu  = s * (1.f / DINNER);
    float var = s2 * (1.f / DINNER) - mu * mu;
    float inv = rsqrtf(var + 1e-5f);
#pragma unroll
    for (int i = 0; i < DINNER / 32; ++i) {
        int dd = lane + i * 32;
        float v  = (vals[i] - mu) * inv * g[dd] + bta[dd];
        float zv = xz[(size_t)row * (2 * DINNER) + DINNER + dd];
        yg[(size_t)row * DINNER + dd] = v * (zv * sigmoidf_(zv));
    }
}

// ---------------------------------------------------------------------------
extern "C" void kernel_launch(void* const* d_in, const int* in_sizes, int n_in,
                              void* d_out, int out_size, void* d_ws, size_t ws_size,
                              hipStream_t stream) {
    const float* x        = (const float*)d_in[0];
    const float* prompt   = (const float*)d_in[1];
    const float* in_w     = (const float*)d_in[2];
    const float* conv_w   = (const float*)d_in[3];
    const float* conv_b   = (const float*)d_in[4];
    const float* xproj_w  = (const float*)d_in[5];
    const float* dtproj_w = (const float*)d_in[6];
    const float* dtproj_b = (const float*)d_in[7];
    const float* A_log    = (const float*)d_in[8];
    const float* Dp       = (const float*)d_in[9];
    const float* pproj_w  = (const float*)d_in[10];
    const float* ln_g     = (const float*)d_in[11];
    const float* ln_b     = (const float*)d_in[12];
    const float* out_w    = (const float*)d_in[13];
    float* out = (float*)d_out;

    // workspace layout (floats)
    float* ws = (float*)d_ws;
    size_t o = 0;
    float* xz   = ws + o; o += (size_t)NROWS * 2 * DINNER;   // 8192x768
    float* xseq = ws + o; o += (size_t)NROWS * DINNER;       // 8192x384
    float* xdbl = ws + o; o += (size_t)NROWS * XDBL_W;       // 8192x44
    float* dtmm = ws + o; o += (size_t)NROWS * DINNER;       // 8192x384
    float* pc   = ws + o; o += (size_t)NROWS * DSTATE;       // 8192x16
    float* ybuf = ws + o; o += (size_t)NROWS * DINNER;       // 8192x384
    float* yg   = ws + o; o += (size_t)NROWS * DINNER;       // 8192x384
    float* Pb   = ws + o; o += (size_t)Bz * DINNER * DSTATE * NCH;
    float* Qb   = ws + o; o += (size_t)Bz * DINNER * DSTATE * NCH;
    float* Hin  = ws + o; o += (size_t)Bz * DINNER * DSTATE * NCH;

    dim3 blk(256);
    auto gemm_grid = [](int M, int N) {
        return dim3((N + BN - 1) / BN, (M + BM - 1) / BM);
    };

    // 1) xz = x @ in_proj_w.T            (8192 x 768, K=192)
    gemm_bf16_wmma<<<gemm_grid(NROWS, 2 * DINNER), blk, 0, stream>>>(
        x, DMODEL, in_w, DMODEL, xz, 2 * DINNER, NROWS, 2 * DINNER, DMODEL);

    // 2) depthwise conv + SiLU -> x_seq
    conv_silu_kernel<<<(NROWS * DINNER + 255) / 256, blk, 0, stream>>>(
        xz, conv_w, conv_b, xseq);

    // 3) x_dbl = x_seq @ x_proj_w.T      (8192 x 44, K=384)
    gemm_bf16_wmma<<<gemm_grid(NROWS, XDBL_W), blk, 0, stream>>>(
        xseq, DINNER, xproj_w, DINNER, xdbl, XDBL_W, NROWS, XDBL_W, DINNER);

    // 4) dt_mm = dt_low @ dt_proj_w.T    (8192 x 384, K=12; A = x_dbl cols 0..11)
    gemm_bf16_wmma<<<gemm_grid(NROWS, DINNER), blk, 0, stream>>>(
        xdbl, XDBL_W, dtproj_w, DTRANK, dtmm, DINNER, NROWS, DINNER, DTRANK);

    // 5) prompt_c = prompt @ prompt_proj_w.T  (8192 x 16, K=192)
    gemm_bf16_wmma<<<gemm_grid(NROWS, DSTATE), blk, 0, stream>>>(
        prompt, DMODEL, pproj_w, DMODEL, pc, DSTATE, NROWS, DSTATE, DMODEL);

    // 6) chunked selective scan
    int scan_threads = Bz * DINNER * DSTATE * NCH;          // 786432
    scan_chunk_kernel<<<scan_threads / 256, blk, 0, stream>>>(
        dtmm, xseq, xdbl, dtproj_b, A_log, Pb, Qb);
    scan_prefix_kernel<<<(Bz * DINNER * DSTATE + 255) / 256, blk, 0, stream>>>(
        Pb, Qb, Hin);
    scan_emit_kernel<<<scan_threads / 256, blk, 0, stream>>>(
        dtmm, xseq, xdbl, pc, dtproj_b, A_log, Dp, Hin, ybuf);

    // 7) LayerNorm + SiLU(z) gate
    ln_gate_kernel<<<NROWS / 8, blk, 0, stream>>>(ybuf, xz, ln_g, ln_b, yg);

    // 8) out = yg @ out_proj_w.T          (8192 x 192, K=384)
    gemm_bf16_wmma<<<gemm_grid(NROWS, DMODEL), blk, 0, stream>>>(
        yg, DINNER, out_w, DINNER, out, DMODEL, NROWS, DMODEL, DINNER);
}
